// SoftMoELayerWrapper_2972117369069
// MI455X (gfx1250) — compile-verified, matrix operand order checked by reading, not run on hardware
//
#include <hip/hip_runtime.h>
#include <hip/hip_bf16.h>
#include <math.h>

#define B_  4
#define M_  8192
#define D_  512
#define N_  128
#define BM_ (B_ * M_)

typedef __attribute__((ext_vector_type(16))) __bf16 v16bf;
typedef __attribute__((ext_vector_type(8)))  float  v8f;

struct U4 { unsigned int x, y, z, w; };
union Frag { v16bf v; U4 q[2]; };

// float -> bf16 (round-to-nearest-even), stored as raw u16
__device__ __forceinline__ unsigned short f2bf(float f) {
  unsigned int u = __float_as_uint(f);
  unsigned int r = (u + 0x7FFFu + ((u >> 16) & 1u)) >> 16;
  return (unsigned short)r;
}

// Load one 16x32 bf16 WMMA fragment from k-contiguous storage:
//   A operand: base[row][k], row0 = m-tile origin
//   B operand: base[col][k], row0 = n-tile origin (B stored column-major-in-n, k-contiguous)
// gfx1250 layout: lanes 0-15 -> row/col = lane, k = k0..k0+7 and k0+16..k0+23
//                 lanes 16-31 -> row/col = lane-16, k offset +8
__device__ __forceinline__ v16bf load_frag_g(const unsigned short* base, int row0, int k0, int ld) {
  const int lane = threadIdx.x & 31;
  const int r    = lane & 15;
  const int kb   = k0 + ((lane & 16) ? 8 : 0);
  const unsigned short* p = base + (size_t)(row0 + r) * (size_t)ld + kb;
  Frag f;
  f.q[0] = *(const U4*)p;
  f.q[1] = *(const U4*)(p + 16);
  return f.v;
}

// ---------------- normalization ----------------
__global__ void k_norm_x(const float* __restrict__ x, unsigned short* __restrict__ xn) {
  const int row = blockIdx.x;              // 0..BM_-1
  const int tid = threadIdx.x;             // 256 threads, 2 floats each
  const float2 v = *(const float2*)(x + (size_t)row * D_ + tid * 2);
  __shared__ float red[256];
  red[tid] = v.x * v.x + v.y * v.y;
  __syncthreads();
  for (int s = 128; s > 0; s >>= 1) { if (tid < s) red[tid] += red[tid + s]; __syncthreads(); }
  const float inv = 1.0f / fmaxf(sqrtf(red[0]), 1e-12f);
  unsigned short* o = xn + (size_t)row * D_ + tid * 2;
  o[0] = f2bf(v.x * inv);
  o[1] = f2bf(v.y * inv);
}

__global__ void k_norm_phi(const float* __restrict__ phi, const float* __restrict__ scale,
                           unsigned short* __restrict__ phin) {
  const int n   = blockIdx.x;              // 0..N_-1
  const int tid = threadIdx.x;             // 256 threads, 2 d's each
  const float v0 = phi[(size_t)tid * N_ + n];
  const float v1 = phi[(size_t)(tid + 256) * N_ + n];
  __shared__ float red[256];
  red[tid] = v0 * v0 + v1 * v1;
  __syncthreads();
  for (int s = 128; s > 0; s >>= 1) { if (tid < s) red[tid] += red[tid + s]; __syncthreads(); }
  const float sc = scale[0] / fmaxf(sqrtf(red[0]), 1e-12f);
  phin[(size_t)n * D_ + tid]       = f2bf(v0 * sc);   // [n][d] : k-contiguous B operand
  phin[(size_t)n * D_ + tid + 256] = f2bf(v1 * sc);
}

// ---------------- GEMM1: logits[bm][n] = xn[bm][d] * phin[n][d]^T ----------------
__global__ void k_logits(const unsigned short* __restrict__ xn, const unsigned short* __restrict__ phin,
                         float* __restrict__ logits) {
  const int wid = blockIdx.x * (blockDim.x >> 5) + (threadIdx.x >> 5);
  const int m0 = (wid >> 3) * 16;          // 2048 m-tiles
  const int n0 = (wid & 7) * 16;           // 8 n-tiles
  v8f acc = {0.f, 0.f, 0.f, 0.f, 0.f, 0.f, 0.f, 0.f};
  for (int k0 = 0; k0 < D_; k0 += 32) {
    v16bf a  = load_frag_g(xn,   m0, k0, D_);
    v16bf bb = load_frag_g(phin, n0, k0, D_);
    acc = __builtin_amdgcn_wmma_f32_16x16x32_bf16(false, a, false, bb, (short)0, acc, false, false);
  }
  const int lane = threadIdx.x & 31;
  const int col  = lane & 15;
  const int ro   = (lane >> 4) << 3;
  for (int r = 0; r < 8; ++r)
    logits[(size_t)(m0 + ro + r) * N_ + (n0 + col)] = acc[r];
}

// ---------------- dispatch softmax over M (column), write transposed dT[b][n][m] ----------------
__global__ void k_softmax_d(const float* __restrict__ logits, unsigned short* __restrict__ dT) {
  const int b   = blockIdx.x / N_;
  const int n   = blockIdx.x % N_;
  const int tid = threadIdx.x;             // 256
  __shared__ float buf[M_];                // 32 KB column cache in LDS
  __shared__ float red[256];
  for (int m = tid; m < M_; m += 256)
    buf[m] = logits[((size_t)b * M_ + m) * N_ + n];
  __syncthreads();
  float mx = -1e30f;
  for (int m = tid; m < M_; m += 256) mx = fmaxf(mx, buf[m]);
  red[tid] = mx; __syncthreads();
  for (int s = 128; s > 0; s >>= 1) { if (tid < s) red[tid] = fmaxf(red[tid], red[tid + s]); __syncthreads(); }
  mx = red[0]; __syncthreads();
  float sum = 0.f;
  for (int m = tid; m < M_; m += 256) sum += __expf(buf[m] - mx);
  red[tid] = sum; __syncthreads();
  for (int s = 128; s > 0; s >>= 1) { if (tid < s) red[tid] += red[tid + s]; __syncthreads(); }
  const float inv = 1.0f / red[0];
  unsigned short* o = dT + ((size_t)b * N_ + n) * M_;
  for (int m = tid; m < M_; m += 256) o[m] = f2bf(__expf(buf[m] - mx) * inv);
}

// ---------------- combine softmax over N (row) ----------------
__global__ void k_softmax_c(const float* __restrict__ logits, unsigned short* __restrict__ c) {
  const int row = blockIdx.x;              // 0..BM_-1
  const int tid = threadIdx.x;             // 128
  const float v = logits[(size_t)row * N_ + tid];
  __shared__ float red[128];
  red[tid] = v; __syncthreads();
  for (int s = 64; s > 0; s >>= 1) { if (tid < s) red[tid] = fmaxf(red[tid], red[tid + s]); __syncthreads(); }
  const float mx = red[0]; __syncthreads();
  const float e = __expf(v - mx);
  red[tid] = e; __syncthreads();
  for (int s = 64; s > 0; s >>= 1) { if (tid < s) red[tid] += red[tid + s]; __syncthreads(); }
  c[(size_t)row * N_ + tid] = f2bf(e / red[0]);
}

// ---------------- LDS tile transpose: xn[b*M+m][e] -> xnT[b][e][m] ----------------
__global__ void k_transpose(const unsigned short* __restrict__ xn, unsigned short* __restrict__ xnT) {
  __shared__ unsigned short t[32][33];
  const int b  = blockIdx.z;
  const int m0 = blockIdx.x * 32;
  const int e0 = blockIdx.y * 32;
  const int tx = threadIdx.x, ty = threadIdx.y;   // (32,8)
  for (int i = 0; i < 4; ++i)
    t[ty + i * 8][tx] = xn[((size_t)b * M_ + m0 + ty + i * 8) * D_ + e0 + tx];
  __syncthreads();
  for (int i = 0; i < 4; ++i)
    xnT[((size_t)b * D_ + e0 + ty + i * 8) * M_ + m0 + tx] = t[tx][ty + i * 8];
}

// ---------------- GEMM3: xs[b][n][e] = sum_m dT[b][n][m] * xnT[b][e][m] ----------------
__global__ void k_xs(const unsigned short* __restrict__ dT, const unsigned short* __restrict__ xnT,
                     unsigned short* __restrict__ xs) {
  const int wid = blockIdx.x * (blockDim.x >> 5) + (threadIdx.x >> 5);
  const int b   = wid >> 8;                 // 256 tiles per batch
  const int rem = wid & 255;
  const int n0  = (rem >> 5) * 16;          // 8 n-tiles
  const int e0  = (rem & 31) * 16;          // 32 e-tiles
  const unsigned short* A  = dT  + (size_t)b * N_ * M_;
  const unsigned short* Bm = xnT + (size_t)b * D_ * M_;
  v8f acc = {0.f, 0.f, 0.f, 0.f, 0.f, 0.f, 0.f, 0.f};
  for (int k0 = 0; k0 < M_; k0 += 32) {     // 256 WMMAs
    v16bf a  = load_frag_g(A,  n0, k0, M_);
    v16bf bb = load_frag_g(Bm, e0, k0, M_);
    acc = __builtin_amdgcn_wmma_f32_16x16x32_bf16(false, a, false, bb, (short)0, acc, false, false);
  }
  const int lane = threadIdx.x & 31;
  const int col  = lane & 15;
  const int ro   = (lane >> 4) << 3;
  for (int r = 0; r < 8; ++r)
    xs[((size_t)b * N_ + (n0 + ro + r)) * D_ + e0 + col] = f2bf(acc[r]);
}

// ---------------- GEMM2: ysT[b][e][n] = sum_d xs[b][n][d] * W[n][d][e] + bias[n][e] ----------------
// W tiles are transposed through LDS so B fragments become contiguous 16B ds loads.
__global__ void k_expert(const unsigned short* __restrict__ xs, const float* __restrict__ W,
                         const float* __restrict__ bias, unsigned short* __restrict__ ysT) {
  const int n     = blockIdx.x >> 2;        // expert
  const int strip = blockIdx.x & 3;         // 128-wide e strip
  const int w     = threadIdx.x >> 5;       // wave in block (8)
  const int lane  = threadIdx.x & 31;
  const int e0    = strip * 128 + w * 16;
  __shared__ unsigned short wt[8][16 * 32]; // per-wave [col][k] bf16 tile of W^T
  const int col = lane & 15;
  const int kh  = (lane >> 4) * 16;
  const unsigned short* A = xs + (size_t)n * D_;   // rows = b (padded to 16), ld = N_*D_
  v8f acc = {0.f, 0.f, 0.f, 0.f, 0.f, 0.f, 0.f, 0.f};
  for (int k0 = 0; k0 < D_; k0 += 32) {
    for (int i = 0; i < 16; ++i) {
      const int k = kh + i;
      wt[w][col * 32 + k] = f2bf(W[((size_t)n * D_ + k0 + k) * D_ + e0 + col]);
    }
    __syncthreads();
    v16bf a = load_frag_g(A, 0, k0, N_ * D_);
    Frag fb;
    const unsigned short* p = &wt[w][col * 32 + ((lane & 16) ? 8 : 0)];
    fb.q[0] = *(const U4*)p;
    fb.q[1] = *(const U4*)(p + 16);
    acc = __builtin_amdgcn_wmma_f32_16x16x32_bf16(false, a, false, fb.v, (short)0, acc, false, false);
    __syncthreads();
  }
  if (lane < 16) {                          // only b = 0..3 rows are valid (vgprs 0..3, lanes 0..15)
    for (int r = 0; r < B_; ++r) {
      const float v = acc[r] + bias[(size_t)n * D_ + e0 + col];
      ysT[((size_t)r * D_ + (e0 + col)) * N_ + n] = f2bf(v);
    }
  }
}

// ---------------- GEMM4: y[b][m][e] = sum_n c[b*M+m][n] * ysT[b][e][n] ----------------
__global__ void k_y(const unsigned short* __restrict__ c, const unsigned short* __restrict__ ysT,
                    float* __restrict__ y) {
  const int wid = blockIdx.x * (blockDim.x >> 5) + (threadIdx.x >> 5);
  const int b   = wid >> 14;                // 16384 tiles per batch
  const int rem = wid & 16383;
  const int m0  = (rem >> 5) * 16;          // 512 m-tiles
  const int e0  = (rem & 31) * 16;          // 32 e-tiles
  const unsigned short* A  = c   + (size_t)b * M_ * N_;
  const unsigned short* Bm = ysT + (size_t)b * D_ * N_;
  v8f acc = {0.f, 0.f, 0.f, 0.f, 0.f, 0.f, 0.f, 0.f};
  for (int k0 = 0; k0 < N_; k0 += 32) {
    v16bf a  = load_frag_g(A,  m0, k0, N_);
    v16bf bb = load_frag_g(Bm, e0, k0, N_);
    acc = __builtin_amdgcn_wmma_f32_16x16x32_bf16(false, a, false, bb, (short)0, acc, false, false);
  }
  const int lane = threadIdx.x & 31;
  const int col  = lane & 15;
  const int ro   = (lane >> 4) << 3;
  for (int r = 0; r < 8; ++r)
    y[((size_t)b * M_ + (m0 + ro + r)) * D_ + (e0 + col)] = acc[r];
}

extern "C" void kernel_launch(void* const* d_in, const int* in_sizes, int n_in,
                              void* d_out, int out_size, void* d_ws, size_t ws_size,
                              hipStream_t stream) {
  (void)in_sizes; (void)n_in; (void)out_size; (void)ws_size;
  const float* x     = (const float*)d_in[0];   // [B,M,D]
  const float* phi   = (const float*)d_in[1];   // [D,N,1]
  const float* scale = (const float*)d_in[2];   // [1]
  const float* W     = (const float*)d_in[3];   // [N,D,D]
  const float* bias  = (const float*)d_in[4];   // [N,D]
  float* y = (float*)d_out;                     // [B,M,D] fp32

  char* ws = (char*)d_ws;
  size_t off = 0;
  auto wsalloc = [&](size_t bytes) -> void* {
    void* p = ws + off;
    off = (off + bytes + 255) & ~(size_t)255;
    return p;
  };
  unsigned short* xn     = (unsigned short*)wsalloc((size_t)BM_ * D_ * 2);   // 32 MB  [bm][d]
  unsigned short* xnT    = (unsigned short*)wsalloc((size_t)B_ * D_ * M_ * 2); // 32 MB [b][e][m]
  unsigned short* phin   = (unsigned short*)wsalloc((size_t)N_ * D_ * 2);    // 128 KB [n][d]
  float*          logits = (float*)wsalloc((size_t)BM_ * N_ * 4);            // 16 MB  [bm][n]
  unsigned short* dT     = (unsigned short*)wsalloc((size_t)B_ * N_ * M_ * 2); // 8 MB [b][n][m]
  unsigned short* cmb    = (unsigned short*)wsalloc((size_t)BM_ * N_ * 2);   // 8 MB   [bm][n]
  unsigned short* xs     = (unsigned short*)wsalloc((size_t)16 * N_ * D_ * 2); // 2 MB (b padded to 16)
  unsigned short* ysT    = (unsigned short*)wsalloc((size_t)B_ * D_ * N_ * 2); // 512 KB [b][e][n]

  hipLaunchKernelGGL(k_norm_x,    dim3(BM_),                  dim3(256),   0, stream, x, xn);
  hipLaunchKernelGGL(k_norm_phi,  dim3(N_),                   dim3(256),   0, stream, phi, scale, phin);
  hipLaunchKernelGGL(k_logits,    dim3(2048),                 dim3(256),   0, stream, xn, phin, logits);
  hipLaunchKernelGGL(k_softmax_d, dim3(B_ * N_),              dim3(256),   0, stream, logits, dT);
  hipLaunchKernelGGL(k_softmax_c, dim3(BM_),                  dim3(128),   0, stream, logits, cmb);
  hipLaunchKernelGGL(k_transpose, dim3(M_ / 32, D_ / 32, B_), dim3(32, 8), 0, stream, xn, xnT);
  hipLaunchKernelGGL(k_xs,        dim3(128),                  dim3(256),   0, stream, dT, xnT, xs);
  hipLaunchKernelGGL(k_expert,    dim3(N_ * 4),               dim3(256),   0, stream, xs, W, bias, ysT);
  hipLaunchKernelGGL(k_y,         dim3(8192),                 dim3(256),   0, stream, cmb, ysT, y);
}
